// VectorQuantizer_43456479101592
// MI455X (gfx1250) — matrix operand order repaired and checked by reading, not compiled, not run on hardware
//
#include <hip/hip_runtime.h>
#include <hip/hip_bf16.h>

// ---------------------------------------------------------------------------
// VQ-VAE vector quantizer forward for MI455X (gfx1250, wave32, WMMA).
// Distance GEMM via v_wmma_f32_16x16x32_f16 with hi/lo f16 split for ~f32
// accuracy; argmin fused in-register; gather + loss in follow-up kernels.
// ---------------------------------------------------------------------------

typedef __attribute__((ext_vector_type(16))) _Float16 v16h;
typedef __attribute__((ext_vector_type(2)))  _Float16 h2;
typedef __attribute__((ext_vector_type(8)))  float    v8f;

#define K_CODES        1024
#define DIM            64
#define N_ROWS         65536          // B*H*W = 64*32*32
#define ROWS_PER_BLOCK 128
#define CODES_PER_TILE 128
#define ZQ_ELEMS       4194304        // B*D*H*W

// ---------------------------------------------------------------------------
// Kernel 1: split codebook rows into f16 hi/lo halves + exact f32 ||e||^2.
// ---------------------------------------------------------------------------
__global__ __launch_bounds__(64) void vq_prep_kernel(
    const float* __restrict__ codebook,
    _Float16* __restrict__ ehi, _Float16* __restrict__ elo,
    float* __restrict__ enorm)
{
    const int k = blockIdx.x;
    const int d = threadIdx.x;
    const float v = codebook[k * DIM + d];
    const _Float16 h = (_Float16)v;
    const _Float16 l = (_Float16)(v - (float)h);
    ehi[k * DIM + d] = h;
    elo[k * DIM + d] = l;

    __shared__ float red[64];
    red[d] = v * v;
    __syncthreads();
    for (int st = 32; st > 0; st >>= 1) {
        if (d < st) red[d] += red[d + st];
        __syncthreads();
    }
    if (d == 0) enorm[k] = red[0];
}

// ---------------------------------------------------------------------------
// Kernel 2: WMMA distance GEMM + fused argmin.
//   dist(n,k) = ||e_k||^2 - 2 * z_n . e_k   (||z||^2 constant per row)
//   z.e computed as zh.eh + zh.el + zl.eh with f32 accumulation (WMMA).
// Block: 256 threads = 8 waves; 128 rows/block; codebook tiled 128 at a time.
// ---------------------------------------------------------------------------
__global__ __launch_bounds__(256) void vq_argmin_kernel(
    const float*    __restrict__ z,
    const _Float16* __restrict__ ehi,
    const _Float16* __restrict__ elo,
    const float*    __restrict__ enorm,
    int*            __restrict__ idx_out,
    float*          __restrict__ idx_f_out)
{
    // Padding chosen so the 16-lane fragment reads hit distinct LDS banks.
    __shared__ _Float16 zsh[ROWS_PER_BLOCK][66];
    __shared__ _Float16 zsl[ROWS_PER_BLOCK][66];
    __shared__ _Float16 esh[CODES_PER_TILE][72];
    __shared__ _Float16 esl[CODES_PER_TILE][72];
    __shared__ float    en [CODES_PER_TILE];

    const int tid   = threadIdx.x;
    const int nbase = blockIdx.x * ROWS_PER_BLOCK;   // multiple of 128 -> one batch b
    const int b     = nbase >> 10;                    // /1024 (H*W)
    const int hw0   = nbase & 1023;
    const float* zb = z + (size_t)b * (DIM * 1024) + hw0;

    // Stage this block's 128 z-rows: memory layout is [b][d][hw] so contiguous
    // along hw (= our row index) -> coalesced 128-float runs per d.
    for (int e = tid; e < ROWS_PER_BLOCK * DIM; e += 256) {
        const int d = e >> 7;       // /128
        const int r = e & 127;
        const float v = zb[(size_t)d * 1024 + r];
        const _Float16 h = (_Float16)v;
        zsh[r][d] = h;
        zsl[r][d] = (_Float16)(v - (float)h);
    }
    __syncthreads();

    const int wave    = tid >> 5;
    const int wl      = tid & 31;
    const int halfsel = wl >> 4;    // wave32 half (lanes 0-15 vs 16-31)
    const int lcol    = wl & 15;
    const int m0      = wave * 16;  // this wave's row tile within the block

    // A fragments (16x32 f16 layout): loop-invariant, kept in VGPRs.
    v16h ah0, ah1, al0, al1;
    {
        const int m = m0 + lcol;
        #pragma unroll
        for (int j = 0; j < 8; ++j) {
            const int kb = ((j >> 2) << 4) + (halfsel << 3) + ((j & 3) << 1);
            const h2 p0 = *(const h2*)&zsh[m][kb];
            const h2 p1 = *(const h2*)&zsh[m][32 + kb];
            const h2 q0 = *(const h2*)&zsl[m][kb];
            const h2 q1 = *(const h2*)&zsl[m][32 + kb];
            ah0[2 * j] = p0[0]; ah0[2 * j + 1] = p0[1];
            ah1[2 * j] = p1[0]; ah1[2 * j + 1] = p1[1];
            al0[2 * j] = q0[0]; al0[2 * j + 1] = q0[1];
            al1[2 * j] = q1[0]; al1[2 * j + 1] = q1[1];
        }
    }

    float bestD[8];
    int   bestI[8];
    #pragma unroll
    for (int v = 0; v < 8; ++v) { bestD[v] = 3.4e38f; bestI[v] = 0; }

    for (int tile = 0; tile < K_CODES / CODES_PER_TILE; ++tile) {
        __syncthreads();
        const int kb0 = tile * CODES_PER_TILE;
        // Stage codebook tile (f16 hi/lo) -- coalesced contiguous copies.
        for (int e = tid; e < CODES_PER_TILE * DIM; e += 256) {
            const int r = e >> 6;      // /DIM
            const int d = e & 63;
            esh[r][d] = ehi[(size_t)(kb0 + r) * DIM + d];
            esl[r][d] = elo[(size_t)(kb0 + r) * DIM + d];
        }
        if (tid < CODES_PER_TILE) en[tid] = enorm[kb0 + tid];
        __syncthreads();

        #pragma unroll
        for (int c = 0; c < CODES_PER_TILE / 16; ++c) {
            const int n = c * 16 + lcol;
            // B fragments (32x16 f16 layout): lane half selects K group.
            v16h bh0, bh1, bl0, bl1;
            #pragma unroll
            for (int i = 0; i < 16; ++i) {
                const int d0 = ((i >> 3) << 4) + (halfsel << 3) + (i & 7);
                bh0[i] = esh[n][d0];
                bh1[i] = esh[n][32 + d0];
                bl0[i] = esl[n][d0];
                bl1[i] = esl[n][32 + d0];
            }

            v8f acc = {};
            // k-step 0 (d = 0..31): compensated product zh.eh + zh.el + zl.eh
            acc = __builtin_amdgcn_wmma_f32_16x16x32_f16(false, ah0, false, bh0,
                                                         (short)0, acc, false, false);
            acc = __builtin_amdgcn_wmma_f32_16x16x32_f16(false, ah0, false, bl0,
                                                         (short)0, acc, false, false);
            acc = __builtin_amdgcn_wmma_f32_16x16x32_f16(false, al0, false, bh0,
                                                         (short)0, acc, false, false);
            // k-step 1 (d = 32..63)
            acc = __builtin_amdgcn_wmma_f32_16x16x32_f16(false, ah1, false, bh1,
                                                         (short)0, acc, false, false);
            acc = __builtin_amdgcn_wmma_f32_16x16x32_f16(false, ah1, false, bl1,
                                                         (short)0, acc, false, false);
            acc = __builtin_amdgcn_wmma_f32_16x16x32_f16(false, al1, false, bh1,
                                                         (short)0, acc, false, false);

            const float env  = en[c * 16 + lcol];
            const int   code = kb0 + c * 16 + lcol;
            #pragma unroll
            for (int v = 0; v < 8; ++v) {
                const float dist = env - 2.0f * acc[v];
                if (dist < bestD[v]) { bestD[v] = dist; bestI[v] = code; }
            }
        }
    }

    // Cross-lane argmin within each wave32 half (16 lanes share a row set):
    // C layout: VGPR v, lanes 0-15 -> row m0+v; lanes 16-31 -> row m0+8+v.
    #pragma unroll
    for (int v = 0; v < 8; ++v) {
        float d = bestD[v];
        int   i = bestI[v];
        #pragma unroll
        for (int off = 8; off > 0; off >>= 1) {
            const float od = __shfl_xor(d, off, 16);
            const int   oi = __shfl_xor(i, off, 16);
            if (od < d || (od == d && oi < i)) { d = od; i = oi; }
        }
        if (lcol == 0) {
            const int row = nbase + m0 + halfsel * 8 + v;
            idx_out[row]   = i;
            idx_f_out[row] = (float)i;
        }
    }
}

// ---------------------------------------------------------------------------
// Kernel 3: gather z_q = codebook[idx] into [B,D,H,W] + per-block partial
// sums of (z_q - z)^2 (deterministic fixed-shape grid-stride loop).
// ---------------------------------------------------------------------------
__global__ __launch_bounds__(256) void vq_gather_kernel(
    const float* __restrict__ z,
    const float* __restrict__ codebook,
    const int*   __restrict__ idx,
    float*       __restrict__ zq_out,
    float*       __restrict__ partials)
{
    float s = 0.0f;
    for (int i = blockIdx.x * 256 + threadIdx.x; i < ZQ_ELEMS;
         i += gridDim.x * 256) {
        const int hw = i & 1023;
        const int bd = i >> 10;
        const int d  = bd & 63;
        const int bb = bd >> 6;
        const int n  = (bb << 10) + hw;
        const int k  = idx[n];
        const float q  = codebook[k * DIM + d];   // codebook resident in L2
        const float zv = z[i];
        zq_out[i] = q;
        const float df = q - zv;
        s += df * df;
    }
    __shared__ float red[256];
    red[threadIdx.x] = s;
    __syncthreads();
    for (int st = 128; st > 0; st >>= 1) {
        if (threadIdx.x < st) red[threadIdx.x] += red[threadIdx.x + st];
        __syncthreads();
    }
    if (threadIdx.x == 0) partials[blockIdx.x] = red[0];
}

// ---------------------------------------------------------------------------
// Kernel 4: final deterministic loss reduction.
// loss = (1 + 0.25) * mean((z_q - z)^2)   [both terms equal numerically]
// ---------------------------------------------------------------------------
__global__ __launch_bounds__(256) void vq_loss_kernel(
    const float* __restrict__ partials, int nparts,
    float* __restrict__ loss_out)
{
    __shared__ float red[256];
    float s = 0.0f;
    for (int i = threadIdx.x; i < nparts; i += 256) s += partials[i];
    red[threadIdx.x] = s;
    __syncthreads();
    for (int st = 128; st > 0; st >>= 1) {
        if (threadIdx.x < st) red[threadIdx.x] += red[threadIdx.x + st];
        __syncthreads();
    }
    if (threadIdx.x == 0)
        loss_out[0] = 1.25f * red[0] * (1.0f / (float)ZQ_ELEMS);
}

// ---------------------------------------------------------------------------
extern "C" void kernel_launch(void* const* d_in, const int* in_sizes, int n_in,
                              void* d_out, int out_size, void* d_ws, size_t ws_size,
                              hipStream_t stream) {
    const float* z        = (const float*)d_in[0];   // [64,64,32,32]
    const float* codebook = (const float*)d_in[1];   // [1024,64]

    // Workspace layout
    char* ws = (char*)d_ws;
    _Float16* ehi   = (_Float16*)(ws + 0);                  // 1024*64*2 = 131072
    _Float16* elo   = (_Float16*)(ws + 131072);             // 131072
    float*    enorm = (float*)(ws + 262144);                // 1024*4   = 4096
    int*      idxw  = (int*)(ws + 266240);                  // 65536*4  = 262144
    float*    parts = (float*)(ws + 528384);                // 1024*4   = 4096

    // Output layout (flat concat): z_q_out [4194304], loss [1], idx [65536]
    float* out     = (float*)d_out;
    float* zq_out  = out;
    float* loss    = out + ZQ_ELEMS;
    float* idx_f   = out + ZQ_ELEMS + 1;

    vq_prep_kernel<<<K_CODES, 64, 0, stream>>>(codebook, ehi, elo, enorm);

    vq_argmin_kernel<<<N_ROWS / ROWS_PER_BLOCK, 256, 0, stream>>>(
        z, ehi, elo, enorm, idxw, idx_f);

    vq_gather_kernel<<<1024, 256, 0, stream>>>(z, codebook, idxw, zq_out, parts);

    vq_loss_kernel<<<1, 256, 0, stream>>>(parts, 1024, loss);
}
